// TransformerEncoder_48352741819065
// MI455X (gfx1250) — matrix operand order repaired and checked by reading, not compiled, not run on hardware
//
#include <hip/hip_runtime.h>
#include <hip/hip_bf16.h>
#include <math.h>

// ---------------------------------------------------------------------------
// CDNA5 / gfx1250 transformer encoder layer.
// All matmuls use v_wmma_f32_16x16x32_bf16 (bf16 A/B, f32 accumulate).
// Attention is flash-style (no S^2 materialization).
// ---------------------------------------------------------------------------

typedef __attribute__((ext_vector_type(16))) __bf16 v16bf;
typedef __attribute__((ext_vector_type(8)))  float  v8f;

union Frag {
  uint4          q[2];
  v16bf          v;
  unsigned short u[16];
};

// Native bf16 convert (single v_cvt on gfx1250; compiler expands if absent).
__device__ __forceinline__ unsigned short f2bf(float f) {
  return __builtin_bit_cast(unsigned short, (__bf16)f);
}

__device__ __forceinline__ v8f wmma_bf16(const Frag& a, const Frag& b, v8f c) {
  return __builtin_amdgcn_wmma_f32_16x16x32_bf16(
      /*neg_a=*/false, a.v, /*neg_b=*/false, b.v,
      /*c_mod=*/(short)0, c, /*reuse_a=*/false, /*reuse_b=*/false);
}

// ---------------------------------------------------------------------------
// GEMM: C[M,N] = A[M,K] @ W[K,N] + bias, optional ReLU.
// Block tile 128x128, BK=64, 256 threads = 8 waves, wave tile 32x64.
// 16 WMMAs per wave per LDS stage (2 k-steps x 2x4 fragments).
// LDS stride 72 shorts = 144B: 16B aligned; bank stride 36 -> the 16 rows a
// fragment load touches hit 16 distinct bank groups (36*r mod 64 all distinct).
// ---------------------------------------------------------------------------
__global__ __launch_bounds__(256) void gemm_bias_kernel(
    const float* __restrict__ A, const float* __restrict__ W,
    const float* __restrict__ bias, float* __restrict__ C,
    int M, int N, int K, int relu)
{
  constexpr int LD = 72;
  __shared__ unsigned short As[128 * LD];   // [row m][k]   (k = 0..63)
  __shared__ unsigned short Bs[128 * LD];   // [col n][k]   (W transposed in LDS)

  const int tid  = threadIdx.x;
  const int wave = tid >> 5;
  const int lane = tid & 31;
  const int half = lane >> 4;
  const int l16  = lane & 15;
  const int bm   = blockIdx.x * 128;
  const int bn   = blockIdx.y * 128;
  const int wm   = (wave >> 1) * 32;        // 4 wave-rows x 32
  const int wn   = (wave & 1) * 64;         // 2 wave-cols x 64

  v8f c[2][4] = {};

  for (int k0 = 0; k0 < K; k0 += 64) {
    // Stage A tile 128x64 (f32 -> bf16): 2048 float4, 8 per thread.
#pragma unroll
    for (int i = 0; i < 8; ++i) {
      int idx = tid + i * 256;
      int row = idx >> 4;                   // 16 float4 per row
      int col = (idx & 15) << 2;
      float4 f = *(const float4*)(A + (size_t)(bm + row) * K + (k0 + col));
      unsigned short* d = &As[row * LD + col];
      d[0] = f2bf(f.x); d[1] = f2bf(f.y); d[2] = f2bf(f.z); d[3] = f2bf(f.w);
    }
    // Stage W tile 64x128 transposed -> Bs[n][k]: 2048 float4, 8 per thread.
#pragma unroll
    for (int i = 0; i < 8; ++i) {
      int idx = tid + i * 256;
      int kk  = idx >> 5;                   // 32 float4 per k-row
      int nn  = (idx & 31) << 2;
      float4 f = *(const float4*)(W + (size_t)(k0 + kk) * N + (bn + nn));
      Bs[(nn + 0) * LD + kk] = f2bf(f.x);
      Bs[(nn + 1) * LD + kk] = f2bf(f.y);
      Bs[(nn + 2) * LD + kk] = f2bf(f.z);
      Bs[(nn + 3) * LD + kk] = f2bf(f.w);
    }
    __syncthreads();

#pragma unroll
    for (int ks = 0; ks < 2; ++ks) {        // two 16x16x32 k-steps per stage
      Frag a[2], b[4];
#pragma unroll
      for (int t = 0; t < 2; ++t) {         // A frag: row = lane&15, K runs
        const unsigned short* r = &As[(wm + t * 16 + l16) * LD + ks * 32];
        a[t].q[0] = ((const uint4*)r)[half];            // K = 8*half + 0..7
        a[t].q[1] = ((const uint4*)(r + 16))[half];     // K = 16 + 8*half + 0..7
      }
#pragma unroll
      for (int j = 0; j < 4; ++j) {         // B frag: col = lane&15, K = 8*half+0..7, 16+8*half+0..7
        const unsigned short* r = &Bs[(wn + j * 16 + l16) * LD + ks * 32];
        b[j].q[0] = ((const uint4*)r)[half];
        b[j].q[1] = ((const uint4*)(r + 16))[half];
      }
#pragma unroll
      for (int i = 0; i < 2; ++i)
#pragma unroll
        for (int j = 0; j < 4; ++j)
          c[i][j] = wmma_bf16(a[i], b[j], c[i][j]);
    }
    __syncthreads();
  }

  // Epilogue: C/D layout row = r + 8*half, col = lane&15
#pragma unroll
  for (int i = 0; i < 2; ++i)
#pragma unroll
    for (int j = 0; j < 4; ++j) {
      int coln = bn + wn + j * 16 + l16;
      float bv = bias[coln];
#pragma unroll
      for (int r = 0; r < 8; ++r) {
        int row = bm + wm + i * 16 + half * 8 + r;
        float val = c[i][j][r] + bv;
        if (relu) val = fmaxf(val, 0.0f);
        C[(size_t)row * N + coln] = val;
      }
    }
}

// ---------------------------------------------------------------------------
// Flash attention, d=64. q/k/v are [B,S,H,d]; output written in [B,H,S,d]
// order (reproducing the reference's reshape-without-transpose bug).
// 256 threads = 8 waves; each wave owns 16 query rows. 64-key blocks.
// ---------------------------------------------------------------------------
__global__ __launch_bounds__(256) void attn_kernel(
    const float* __restrict__ q, const float* __restrict__ k,
    const float* __restrict__ v, float* __restrict__ o,
    int B, int H, int S)
{
  constexpr int D  = 64;
  constexpr int LD = 72;                     // 144B stride: 16B aligned, conflict-free
  __shared__ unsigned short Kt[64 * LD];     // [key][d]   -> B-frag layout for Q@K^T
  __shared__ unsigned short Vt[64 * LD];     // [d][key]   -> B-frag layout for P@V
  __shared__ unsigned short Ps[8 * 16 * LD]; // per-wave P scratch [16][64 keys]

  const int tid  = threadIdx.x;
  const int wave = tid >> 5;
  const int lane = tid & 31;
  const int half = lane >> 4;
  const int l16  = lane & 15;
  const int qb   = blockIdx.x * 128;
  const int b    = blockIdx.y / H;
  const int h    = blockIdx.y % H;
  const float scale = 0.125f;                // 1/sqrt(64)

  // Q A-fragments held in registers for the whole kernel (pre-scaled).
  const int mrow = qb + wave * 16 + l16;
  const float* qrow = q + (((size_t)b * S + mrow) * H + h) * D;
  Frag aq[2];
#pragma unroll
  for (int ks = 0; ks < 2; ++ks) {
#pragma unroll
    for (int i = 0; i < 8; ++i)
      aq[ks].u[i]     = f2bf(qrow[ks * 32 + half * 8 + i] * scale);
#pragma unroll
    for (int i = 0; i < 8; ++i)
      aq[ks].u[8 + i] = f2bf(qrow[ks * 32 + 16 + half * 8 + i] * scale);
  }

  float m_r[8], l_r[8];
#pragma unroll
  for (int r = 0; r < 8; ++r) { m_r[r] = -__builtin_inff(); l_r[r] = 0.0f; }
  v8f oc[4] = {};

  unsigned short* Pw = &Ps[wave * 16 * LD];

  for (int kb = 0; kb < S / 64; ++kb) {
    // Cooperative K/V staging: thread -> key row tid>>2, 16 d-columns.
    {
      int row = tid >> 2;
      int seg = tid & 3;
      const float* krow = k + (((size_t)b * S + kb * 64 + row) * H + h) * D;
      const float* vrow = v + (((size_t)b * S + kb * 64 + row) * H + h) * D;
#pragma unroll
      for (int j = 0; j < 4; ++j) {
        int cb = seg * 16 + j * 4;
        float4 f = *(const float4*)(krow + cb);
        unsigned short* d = &Kt[row * LD + cb];
        d[0] = f2bf(f.x); d[1] = f2bf(f.y); d[2] = f2bf(f.z); d[3] = f2bf(f.w);
        float4 g = *(const float4*)(vrow + cb);
        Vt[(cb + 0) * LD + row] = f2bf(g.x);
        Vt[(cb + 1) * LD + row] = f2bf(g.y);
        Vt[(cb + 2) * LD + row] = f2bf(g.z);
        Vt[(cb + 3) * LD + row] = f2bf(g.w);
      }
    }
    __syncthreads();

    // Scores: 16 query rows x 64 keys
    v8f sf[4] = {};
#pragma unroll
    for (int ks = 0; ks < 2; ++ks)
#pragma unroll
      for (int nt = 0; nt < 4; ++nt) {
        Frag bk;
        const unsigned short* r = &Kt[(nt * 16 + l16) * LD + ks * 32];
        bk.q[0] = ((const uint4*)r)[half];
        bk.q[1] = ((const uint4*)(r + 16))[half];
        sf[nt] = wmma_bf16(aq[ks], bk, sf[nt]);
      }

    // Online softmax; C-frag element r corresponds to row (r + 8*half),
    // columns live in lanes of the same 16-lane half -> xor{8,4,2,1} reductions.
    float mloc[8], pl[8], alpha[8];
#pragma unroll
    for (int r = 0; r < 8; ++r)
      mloc[r] = fmaxf(fmaxf(sf[0][r], sf[1][r]), fmaxf(sf[2][r], sf[3][r]));
#pragma unroll
    for (int off = 8; off >= 1; off >>= 1)
#pragma unroll
      for (int r = 0; r < 8; ++r)
        mloc[r] = fmaxf(mloc[r], __shfl_xor(mloc[r], off, 32));
#pragma unroll
    for (int r = 0; r < 8; ++r) {
      float mn = fmaxf(m_r[r], mloc[r]);
      alpha[r] = __expf(m_r[r] - mn);
      m_r[r]   = mn;
      pl[r]    = 0.0f;
    }
#pragma unroll
    for (int nt = 0; nt < 4; ++nt)
#pragma unroll
      for (int r = 0; r < 8; ++r) {
        float p = __expf(sf[nt][r] - m_r[r]);
        pl[r] += p;
        Pw[(half * 8 + r) * LD + nt * 16 + l16] = f2bf(p);
      }
#pragma unroll
    for (int off = 8; off >= 1; off >>= 1)
#pragma unroll
      for (int r = 0; r < 8; ++r)
        pl[r] += __shfl_xor(pl[r], off, 32);
#pragma unroll
    for (int r = 0; r < 8; ++r)
      l_r[r] = l_r[r] * alpha[r] + pl[r];
#pragma unroll
    for (int nt = 0; nt < 4; ++nt)
#pragma unroll
      for (int r = 0; r < 8; ++r)
        oc[nt][r] *= alpha[r];

    // O += P @ V   (A-frag re-layout of P via this wave's private LDS scratch)
#pragma unroll
    for (int ks = 0; ks < 2; ++ks) {
      Frag ap;
      const unsigned short* pr = &Pw[l16 * LD + ks * 32];
      ap.q[0] = ((const uint4*)(pr + half * 8))[0];
      ap.q[1] = ((const uint4*)(pr + 16 + half * 8))[0];
#pragma unroll
      for (int nt = 0; nt < 4; ++nt) {
        Frag bv;
        const unsigned short* r = &Vt[(nt * 16 + l16) * LD + ks * 32];
        bv.q[0] = ((const uint4*)r)[half];
        bv.q[1] = ((const uint4*)(r + 16))[half];
        oc[nt] = wmma_bf16(ap, bv, oc[nt]);
      }
    }
    __syncthreads();
  }

  // Write zs in [B,H,S,D] order (== the reference's buggy [B,S,E] flat view).
  float rinv[8];
#pragma unroll
  for (int r = 0; r < 8; ++r) rinv[r] = 1.0f / l_r[r];
#pragma unroll
  for (int nt = 0; nt < 4; ++nt)
#pragma unroll
    for (int r = 0; r < 8; ++r) {
      int srow = qb + wave * 16 + half * 8 + r;
      int dd   = nt * 16 + l16;
      o[(((size_t)b * H + h) * S + srow) * D + dd] = oc[nt][r] * rinv[r];
    }
}

// ---------------------------------------------------------------------------
// out[row] = g * ((a+b) - mu) / sqrt(var + 1e-3) + beta ; one block per row.
// ---------------------------------------------------------------------------
__global__ __launch_bounds__(128) void ln_residual_kernel(
    const float* __restrict__ a, const float* __restrict__ bb,
    const float* __restrict__ g, const float* __restrict__ beta,
    float* __restrict__ out, int E)
{
  const int row  = blockIdx.x;
  const int tid  = threadIdx.x;
  const int lane = tid & 31;
  const int wave = tid >> 5;
  const float* ar = a  + (size_t)row * E;
  const float* br = bb + (size_t)row * E;

  float vbuf[4];
  float s = 0.f, s2 = 0.f;
#pragma unroll
  for (int i = 0; i < 4; ++i) {
    float t = ar[tid + i * 128] + br[tid + i * 128];
    vbuf[i] = t; s += t; s2 += t * t;
  }
#pragma unroll
  for (int off = 16; off >= 1; off >>= 1) {
    s  += __shfl_xor(s,  off, 32);
    s2 += __shfl_xor(s2, off, 32);
  }
  __shared__ float red[8];
  if (lane == 0) { red[wave] = s; red[4 + wave] = s2; }
  __syncthreads();
  s  = red[0] + red[1] + red[2] + red[3];
  s2 = red[4] + red[5] + red[6] + red[7];

  const float invE = 1.0f / (float)E;
  float mu  = s * invE;
  float var = fmaxf(s2 * invE - mu * mu, 0.0f);
  float inv = 1.0f / sqrtf(var + 1e-3f);
#pragma unroll
  for (int i = 0; i < 4; ++i) {
    int col = tid + i * 128;
    out[(size_t)row * E + col] = g[col] * (vbuf[i] - mu) * inv + beta[col];
  }
}

// ---------------------------------------------------------------------------
extern "C" void kernel_launch(void* const* d_in, const int* in_sizes, int n_in,
                              void* d_out, int out_size, void* d_ws, size_t ws_size,
                              hipStream_t stream) {
  (void)in_sizes; (void)n_in; (void)out_size; (void)ws_size;
  constexpr int Bb = 2, S = 4096, E = 512, H = 8, F = 2048;
  constexpr int BS = Bb * S;                 // 8192 rows

  const float* x   = (const float*)d_in[0];
  const float* WQ  = (const float*)d_in[1];
  const float* bQ  = (const float*)d_in[2];
  const float* WK  = (const float*)d_in[3];
  const float* bK  = (const float*)d_in[4];
  const float* WV  = (const float*)d_in[5];
  const float* bV  = (const float*)d_in[6];
  const float* WO  = (const float*)d_in[7];
  const float* bO  = (const float*)d_in[8];
  const float* W1  = (const float*)d_in[9];
  const float* b1  = (const float*)d_in[10];
  const float* W2  = (const float*)d_in[11];
  const float* b2  = (const float*)d_in[12];
  const float* g1  = (const float*)d_in[13];
  const float* be1 = (const float*)d_in[14];
  const float* g2  = (const float*)d_in[15];
  const float* be2 = (const float*)d_in[16];
  float* out = (float*)d_out;

  // Workspace layout (floats): 7 * BS*E + BS*F  (~185 MB)
  float* ws = (float*)d_ws;
  const size_t n_e = (size_t)BS * E;
  float* qb = ws;
  float* kb = qb + n_e;
  float* vb = kb + n_e;
  float* zs = vb + n_e;
  float* t0 = zs + n_e;
  float* z  = t0 + n_e;
  float* a1 = z  + n_e;
  float* r0 = a1 + (size_t)BS * F;

  dim3 blk(256);
  dim3 gEE(BS / 128, E / 128);

  gemm_bias_kernel<<<gEE, blk, 0, stream>>>(x, WQ, bQ, qb, BS, E, E, 0);
  gemm_bias_kernel<<<gEE, blk, 0, stream>>>(x, WK, bK, kb, BS, E, E, 0);
  gemm_bias_kernel<<<gEE, blk, 0, stream>>>(x, WV, bV, vb, BS, E, E, 0);
  attn_kernel<<<dim3(S / 128, Bb * H), blk, 0, stream>>>(qb, kb, vb, zs, Bb, H, S);
  gemm_bias_kernel<<<gEE, blk, 0, stream>>>(zs, WO, bO, t0, BS, E, E, 0);
  ln_residual_kernel<<<dim3(BS), dim3(128), 0, stream>>>(t0, x, g1, be1, z, E);
  gemm_bias_kernel<<<dim3(BS / 128, F / 128), blk, 0, stream>>>(z, W1, b1, a1, BS, F, E, 1);
  gemm_bias_kernel<<<gEE, blk, 0, stream>>>(a1, W2, b2, r0, BS, E, F, 0);
  ln_residual_kernel<<<dim3(BS), dim3(128), 0, stream>>>(r0, z, g2, be2, out, E);
}
